// SelfAttentionV2_13185549598788
// MI455X (gfx1250) — matrix operand order repaired and checked by reading, not compiled, not run on hardware
//
#include <hip/hip_runtime.h>
#include <stdint.h>

#define D_MODEL 1024
#define D_KV    128
#define BATCH   8
#define SEQ     2048
#define SKB     128     // sk block for flash loop (8 tiles of 16)
#define KROW    136     // padded LDS row pitch in ushorts (256B data + 16B pad)

typedef __attribute__((ext_vector_type(16))) __bf16       v16bf;
typedef __attribute__((ext_vector_type(8)))  float        v8f;
typedef __attribute__((ext_vector_type(4)))  unsigned int u32x4;
typedef __attribute__((ext_vector_type(8)))  int          i32x8;
typedef __attribute__((ext_vector_type(4)))  int          i32x4;

union BfFrag {
    v16bf          v;
    __bf16         b[16];
    unsigned short s[16];
    u32x4          q[2];
};

__device__ __forceinline__ unsigned short f2bf_s(float f) {
    __bf16 h = (__bf16)f;                       // fptrunc, RNE -> native cvt
    return __builtin_bit_cast(unsigned short, h);
}

__device__ __forceinline__ v8f wmma_bf16(const v16bf& a, const v16bf& b, const v8f& c) {
    return __builtin_amdgcn_wmma_f32_16x16x32_bf16(false, a, false, b, (short)0, c, false, false);
}

// ---------------------------------------------------------------------------
// TDM: async 2-D tile load global -> LDS (ISA 08_async_tensor §8 D# layout).
// rows x row_bytes tile; 16B LDS padding appended after every 256B of data
// (pad_interval=5 -> 64 DW, pad_amount=3 -> 4 DW) to rotate LDS banks.
// ---------------------------------------------------------------------------
__device__ __forceinline__ void tdm_load_2d(unsigned lds_off, const void* gptr,
                                            unsigned row_bytes, unsigned rows,
                                            unsigned row_stride_bytes) {
    unsigned long long ga = (unsigned long long)(uintptr_t)gptr;
    u32x4 g0;
    g0[0] = 1u;                                              // count=1 (valid), user mode
    g0[1] = lds_off;                                         // lds_addr (bytes)
    g0[2] = (unsigned)(ga & 0xFFFFFFFFu);                    // global_addr[31:0]
    g0[3] = (unsigned)((ga >> 32) & 0x1FFFFFFu) | (2u << 30);// global_addr[56:32] | type=2
    i32x8 g1;
    g1[0] = (int)((1u << 20) | (5u << 22) | (3u << 25));     // pad_enable, interval=64DW, amount=4DW
    g1[1] = (int)((row_bytes & 0xFFFFu) << 16);              // tensor_dim0 lo16
    g1[2] = (int)(((row_bytes >> 16) & 0xFFFFu) | ((rows & 0xFFFFu) << 16)); // dim0 hi | dim1 lo
    g1[3] = (int)(((rows >> 16) & 0xFFFFu) | ((row_bytes & 0xFFFFu) << 16)); // dim1 hi | tile_dim0
    g1[4] = (int)(rows & 0xFFFFu);                           // tile_dim1 | tile_dim2=0
    g1[5] = (int)row_stride_bytes;                           // tensor_dim0_stride lo32
    g1[6] = 0;                                               // stride hi | dim1_stride lo
    g1[7] = 0;
    i32x4 z4 = {0, 0, 0, 0};
#if __clang_major__ >= 23
    i32x8 z8 = {0, 0, 0, 0, 0, 0, 0, 0};
    __builtin_amdgcn_tensor_load_to_lds(g0, g1, z4, z4, z8, 0);
#else
    __builtin_amdgcn_tensor_load_to_lds(g0, g1, z4, z4, 0);
#endif
}

// ---------------------------------------------------------------------------
// Kernel 0: convert W [1024,128] f32 -> Wt [128,1024] bf16 (B-operand layout)
// ---------------------------------------------------------------------------
__global__ void wt_prep_kernel(const float* __restrict__ Wq,
                               const float* __restrict__ Wk,
                               const float* __restrict__ Wv,
                               unsigned short* __restrict__ Wt) {
    int tid = blockIdx.x * blockDim.x + threadIdx.x;
    int z   = tid >> 17;
    int rem = tid & 131071;
    int k   = rem >> 7;
    int n   = rem & 127;
    const float* W = (z == 0) ? Wq : (z == 1) ? Wk : Wv;
    Wt[z * 131072 + n * 1024 + k] = f2bf_s(W[k * 128 + n]);
}

// ---------------------------------------------------------------------------
// Kernel 1: projection  Out = X @ W + b   (bf16 WMMA, f32 accum)
// ---------------------------------------------------------------------------
__global__ __launch_bounds__(128) void proj_kernel(
    const float* __restrict__ X,
    const unsigned short* __restrict__ Wt,
    const float* __restrict__ bias,
    unsigned short* __restrict__ Out,
    int transposed)
{
    const int lane = threadIdx.x & 31;
    const int wave = threadIdx.x >> 5;
    const int m    = lane & 15;
    const int h    = lane >> 4;
    const int row0 = blockIdx.x * 64 + wave * 16;

    v8f acc[8];
    #pragma unroll
    for (int j = 0; j < 8; ++j)
        #pragma unroll
        for (int r = 0; r < 8; ++r) acc[j][r] = 0.0f;

    const float* xrow = X + (size_t)(row0 + m) * D_MODEL;

    for (int kc = 0; kc < D_MODEL; kc += 32) {
        __builtin_prefetch(xrow + kc + 64, 0, 1);

        BfFrag a;
        #pragma unroll
        for (int i = 0; i < 8; ++i) a.b[i]     = (__bf16)xrow[kc + h * 16 + i];
        #pragma unroll
        for (int i = 0; i < 8; ++i) a.b[8 + i] = (__bf16)xrow[kc + h * 16 + 8 + i];

        #pragma unroll
        for (int j = 0; j < 8; j += 2) {
            BfFrag b0, b1;
            const unsigned short* wp0 = Wt + (size_t)(j * 16 + m) * D_MODEL + kc + h * 16;
            const unsigned short* wp1 = wp0 + 16 * D_MODEL;
            b0.q[0] = *(const u32x4*)(wp0); b0.q[1] = *(const u32x4*)(wp0 + 8);
            b1.q[0] = *(const u32x4*)(wp1); b1.q[1] = *(const u32x4*)(wp1 + 8);
            acc[j]     = wmma_bf16(a.v, b0.v, acc[j]);
            acc[j + 1] = wmma_bf16(a.v, b1.v, acc[j + 1]);
        }
    }

    #pragma unroll
    for (int j = 0; j < 8; ++j) {
        float bb = bias[j * 16 + m];
        #pragma unroll
        for (int r = 0; r < 8; ++r) acc[j][r] += bb;
    }

    if (!transposed) {
        #pragma unroll
        for (int j = 0; j < 8; ++j)
            #pragma unroll
            for (int r = 0; r < 8; ++r)
                Out[(size_t)(row0 + 8 * h + r) * D_KV + j * 16 + m] = f2bf_s(acc[j][r]);
    } else {
        int b  = row0 >> 11;
        int s0 = (row0 & (SEQ - 1)) + 8 * h;
        #pragma unroll
        for (int j = 0; j < 8; ++j) {
            u32x4 pk;
            #pragma unroll
            for (int p = 0; p < 4; ++p) {
                unsigned lo = f2bf_s(acc[j][2 * p]);
                unsigned hi = f2bf_s(acc[j][2 * p + 1]);
                pk[p] = lo | (hi << 16);
            }
            unsigned short* vp = Out + (size_t)b * D_KV * SEQ
                                     + (size_t)(j * 16 + m) * SEQ + s0;
            *(u32x4*)vp = pk;
        }
    }
}

// ---------------------------------------------------------------------------
// Kernel 2: flash attention, sk block = 128.
//   K block staged once per workgroup via TDM into padded LDS; V from global.
// ---------------------------------------------------------------------------
__global__ __launch_bounds__(128) void attn_kernel(
    const unsigned short* __restrict__ Qb,
    const unsigned short* __restrict__ Kb,
    const unsigned short* __restrict__ Vt,
    const int* __restrict__ mask,
    float* __restrict__ Out)
{
    __shared__ __align__(16) unsigned short Klds[SKB * KROW];   // 34 KB, padded rows
    __shared__ __align__(16) unsigned short Pbuf[4][16 * SKB];  // 16 KB

    const int lane = threadIdx.x & 31;
    const int wave = threadIdx.x >> 5;
    const int n    = lane & 15;
    const int h    = lane >> 4;
    const int b    = blockIdx.y;
    const int q0   = blockIdx.x * 64 + wave * 16;
    const size_t qrow = (size_t)b * SEQ + q0;

    BfFrag qa[4];
    #pragma unroll
    for (int c = 0; c < 4; ++c) {
        const unsigned short* qp = Qb + (qrow + n) * D_KV + c * 32;
        qa[c].q[0] = *(const u32x4*)(qp + h * 16);
        qa[c].q[1] = *(const u32x4*)(qp + h * 16 + 8);
    }

    v8f   acc[8];
    float mrun[8], lrun[8];
    #pragma unroll
    for (int r = 0; r < 8; ++r) { mrun[r] = -1e30f; lrun[r] = 0.0f; }
    #pragma unroll
    for (int j = 0; j < 8; ++j)
        #pragma unroll
        for (int r = 0; r < 8; ++r) acc[j][r] = 0.0f;

    const float scale = 0.08838834764831845f;  // 1/sqrt(128)
    const unsigned klds_off = (unsigned)(uintptr_t)(&Klds[0]);

    for (int skb = 0; skb < SEQ; skb += SKB) {
        __syncthreads();   // previous iteration's Klds reads complete

        if (wave == 0) {
            // 128 rows x 256B of K[b, skb..skb+127, :] -> Klds (row pitch 272B)
            tdm_load_2d(klds_off,
                        Kb + ((size_t)b * SEQ + skb) * D_KV,
                        /*row_bytes=*/D_KV * 2, /*rows=*/SKB,
                        /*row_stride_bytes=*/D_KV * 2);
        }
        __builtin_amdgcn_s_wait_tensorcnt(0);
        __syncthreads();   // staged K visible to all waves

        __builtin_prefetch(Kb + ((size_t)b * SEQ + skb + SKB + n) * D_KV, 0, 1);

        int   mskv[8];
        float addm[8];
        #pragma unroll
        for (int t = 0; t < 8; ++t) {
            mskv[t] = mask[b * SEQ + skb + 16 * t + n];
            addm[t] = mskv[t] ? 0.0f : -1e9f;
        }

        v8f c[8];
        #pragma unroll
        for (int t = 0; t < 8; ++t)
            #pragma unroll
            for (int r = 0; r < 8; ++r) c[t][r] = 0.0f;

        // scores: 8 tiles x 4 K-chunks = 32 WMMAs, B-frags from LDS
        #pragma unroll
        for (int kc = 0; kc < 4; ++kc) {
            #pragma unroll
            for (int t = 0; t < 8; t += 2) {
                BfFrag k0, k1;
                const unsigned short* kp0 = &Klds[(16 * t + n) * KROW + kc * 32 + h * 16];
                const unsigned short* kp1 = kp0 + 16 * KROW;
                k0.q[0] = *(const u32x4*)(kp0); k0.q[1] = *(const u32x4*)(kp0 + 8);
                k1.q[0] = *(const u32x4*)(kp1); k1.q[1] = *(const u32x4*)(kp1 + 8);
                c[t]     = wmma_bf16(qa[kc].v, k0.v, c[t]);
                c[t + 1] = wmma_bf16(qa[kc].v, k1.v, c[t + 1]);
            }
        }

        // online softmax per row (row m = r + 8h; stats across 16-lane halves)
        #pragma unroll
        for (int r = 0; r < 8; ++r) {
            float s[8];
            #pragma unroll
            for (int t = 0; t < 8; ++t) s[t] = c[t][r] * scale + addm[t];
            float rm = s[0];
            #pragma unroll
            for (int t = 1; t < 8; ++t) rm = fmaxf(rm, s[t]);
            rm = fmaxf(rm, __shfl_xor(rm, 1));
            rm = fmaxf(rm, __shfl_xor(rm, 2));
            rm = fmaxf(rm, __shfl_xor(rm, 4));
            rm = fmaxf(rm, __shfl_xor(rm, 8));
            float mnew  = fmaxf(mrun[r], rm);
            float alpha = __expf(mrun[r] - mnew);
            float p[8], rs = 0.0f;
            #pragma unroll
            for (int t = 0; t < 8; ++t) {
                p[t] = mskv[t] ? __expf(s[t] - mnew) : 0.0f;
                rs += p[t];
            }
            rs += __shfl_xor(rs, 1);
            rs += __shfl_xor(rs, 2);
            rs += __shfl_xor(rs, 4);
            rs += __shfl_xor(rs, 8);
            lrun[r] = lrun[r] * alpha + rs;
            mrun[r] = mnew;
            #pragma unroll
            for (int j = 0; j < 8; ++j) acc[j][r] *= alpha;

            const int mrow = r + 8 * h;
            #pragma unroll
            for (int t = 0; t < 8; ++t)
                Pbuf[wave][mrow * SKB + 16 * t + n] = f2bf_s(p[t]);
        }
        __syncthreads();

        BfFrag pa[4];
        #pragma unroll
        for (int cc = 0; cc < 4; ++cc) {
            pa[cc].q[0] = *(const u32x4*)&Pbuf[wave][n * SKB + cc * 32 + h * 16];
            pa[cc].q[1] = *(const u32x4*)&Pbuf[wave][n * SKB + cc * 32 + h * 16 + 8];
        }

        // acc += P @ V : 4 K-chunks x 8 dv tiles = 32 WMMAs
        #pragma unroll
        for (int cc = 0; cc < 4; ++cc) {
            #pragma unroll
            for (int j = 0; j < 8; j += 2) {
                BfFrag v0, v1;
                const unsigned short* vp0 = Vt + (size_t)b * D_KV * SEQ
                                               + (size_t)(j * 16 + n) * SEQ + skb + cc * 32 + h * 16;
                const unsigned short* vp1 = vp0 + 16 * SEQ;
                v0.q[0] = *(const u32x4*)(vp0); v0.q[1] = *(const u32x4*)(vp0 + 8);
                v1.q[0] = *(const u32x4*)(vp1); v1.q[1] = *(const u32x4*)(vp1 + 8);
                acc[j]     = wmma_bf16(pa[cc].v, v0.v, acc[j]);
                acc[j + 1] = wmma_bf16(pa[cc].v, v1.v, acc[j + 1]);
            }
        }
    }

    #pragma unroll
    for (int r = 0; r < 8; ++r) {
        float inv = (lrun[r] > 0.0f) ? (1.0f / lrun[r]) : 0.0f;
        size_t orow = (qrow + r + 8 * h) * D_KV;
        #pragma unroll
        for (int j = 0; j < 8; ++j)
            Out[orow + j * 16 + n] = acc[j][r] * inv;
    }
}

// ---------------------------------------------------------------------------
extern "C" void kernel_launch(void* const* d_in, const int* in_sizes, int n_in,
                              void* d_out, int out_size, void* d_ws, size_t ws_size,
                              hipStream_t stream) {
    (void)in_sizes; (void)n_in; (void)out_size; (void)ws_size;

    const float* in_q = (const float*)d_in[0];
    const float* in_k = (const float*)d_in[1];
    const float* in_v = (const float*)d_in[2];
    const int*   msk  = (const int*)  d_in[3];
    const float* Wq   = (const float*)d_in[4];
    const float* bq   = (const float*)d_in[5];
    const float* Wk   = (const float*)d_in[6];
    const float* bk   = (const float*)d_in[7];
    const float* Wv   = (const float*)d_in[8];
    const float* bv   = (const float*)d_in[9];

    unsigned short* ws = (unsigned short*)d_ws;
    unsigned short* Wt = ws;
    unsigned short* Qb = Wt + 3 * 131072;
    unsigned short* Kb = Qb + (size_t)BATCH * SEQ * D_KV;
    unsigned short* Vt = Kb + (size_t)BATCH * SEQ * D_KV;

    wt_prep_kernel<<<1536, 256, 0, stream>>>(Wq, Wk, Wv, Wt);

    proj_kernel<<<(BATCH * SEQ) / 64, 128, 0, stream>>>(in_q, Wt,              bq, Qb, 0);
    proj_kernel<<<(BATCH * SEQ) / 64, 128, 0, stream>>>(in_k, Wt + 131072,     bk, Kb, 0);
    proj_kernel<<<(BATCH * SEQ) / 64, 128, 0, stream>>>(in_v, Wt + 2 * 131072, bv, Vt, 1);

    dim3 grid(SEQ / 64, BATCH);
    attn_kernel<<<grid, 128, 0, stream>>>(Qb, Kb, Vt, msk, (float*)d_out);
}
